// DGCNN_50878182588945
// MI455X (gfx1250) — matrix operand order repaired and checked by reading, not compiled, not run on hardware
//
#include <hip/hip_runtime.h>
#include <hip/hip_bf16.h>

// ---------------------------------------------------------------------------
// CDNA5 WMMA helpers (gfx1250, wave32)
// ---------------------------------------------------------------------------
typedef __attribute__((ext_vector_type(16))) _Float16 v16h;
typedef __attribute__((ext_vector_type(8)))  _Float16 v8h;
typedef __attribute__((ext_vector_type(8)))  float    v8f;
typedef __attribute__((ext_vector_type(4)))  float    v4f;

union HalfPack { v4f f[2]; v16h h; };
union RunU     { v4f f;    v8h  h; };

__device__ __forceinline__ v8f wmma16(v16h a, v16h b, v8f c) {
  // D = A(16x32 f16) x B(32x16 f16) + C(16x16 f32)
  return __builtin_amdgcn_wmma_f32_16x16x32_f16(false, a, false, b, (short)0, c,
                                                false, false);
}

// A-operand (16x32 f16): lane m = lane&15, h = lane>>4. Halves are two
// contiguous 8-half (16B) runs at K = h*16 and K = h*16 + 8.
// B-operand (32x16 f16): lane n = lane&15 is the column, halves are
// K = (lane>>4)*16 + q : 16 contiguous halves.
__device__ __forceinline__ v16h load16h(const _Float16* p) {
  HalfPack u;
  const v4f* q = reinterpret_cast<const v4f*>(p);
  u.f[0] = q[0]; u.f[1] = q[1];
  return u.h;
}

__device__ __forceinline__ v16h loadA_contig(const _Float16* row, int kb, int h) {
  HalfPack u;
  u.f[0] = *reinterpret_cast<const v4f*>(row + kb + h * 16);
  u.f[1] = *reinterpret_cast<const v4f*>(row + kb + h * 16 + 8);
  return u.h;
}

// Order-preserving float<->uint mapping (for exact deterministic atomic max).
__device__ __forceinline__ unsigned f2sort(float f) {
  unsigned u = __float_as_uint(f);
  return (u & 0x80000000u) ? ~u : (u | 0x80000000u);
}
__device__ __forceinline__ float sort2f(unsigned s) {
  unsigned u = (s & 0x80000000u) ? (s & 0x7fffffffu) : ~s;
  return __uint_as_float(u);
}

// Build one 8-half run of an edge feature row ef = [nbr-ctr (c), ctr (c), 0...].
// s is a multiple of 8; when c % 8 == 0 each run is region-pure -> packed path.
__device__ __forceinline__ v4f build_ef_run(const _Float16* __restrict__ nbrRow,
                                            const _Float16* __restrict__ ctrRow,
                                            int s, int c, bool vecOK) {
  if (vecOK) {
    if (s + 8 <= c) {                       // diff region: packed f16 subtract
      RunU a, b, d;
      a.f = *reinterpret_cast<const v4f*>(nbrRow + s);
      b.f = *reinterpret_cast<const v4f*>(ctrRow + s);
      d.h = a.h - b.h;
      return d.f;
    }
    if (s >= 2 * c) { v4f z = {0.f, 0.f, 0.f, 0.f}; return z; }
    if (s >= c)     return *reinterpret_cast<const v4f*>(ctrRow + s - c);
  }
  RunU d;                                   // scalar fallback (conv1, c=3)
#pragma unroll
  for (int j = 0; j < 8; j++) {
    int kq = s + j;
    float v;
    if (kq < c)          v = (float)nbrRow[kq] - (float)ctrRow[kq];
    else if (kq < 2 * c) v = (float)ctrRow[kq - c];
    else                 v = 0.f;
    d.h[j] = (_Float16)v;
  }
  return d.f;
}

// ---------------------------------------------------------------------------
// Small data-movement / conversion kernels
// ---------------------------------------------------------------------------
__global__ void extract_pts_kernel(const float* __restrict__ x, float* __restrict__ pts,
                                   int B, int N) {
  int t = blockIdx.x * blockDim.x + threadIdx.x;
  if (t >= B * N * 3) return;
  int b = t / (N * 3);
  int n = (t / 3) % N;
  int d = t % 3;
  pts[t] = x[((size_t)(b * 3 + d)) * N + n];   // x is (B,3,N,1)
}

__global__ void f32_to_f16_pad_kernel(const float* __restrict__ src, _Float16* __restrict__ dst,
                                      int rows, int c, int cp) {
  int t = blockIdx.x * blockDim.x + threadIdx.x;
  if (t >= rows * cp) return;
  int r = t / cp, ch = t % cp;
  dst[t] = (ch < c) ? (_Float16)src[(size_t)r * c + ch] : (_Float16)0.f;
}

__global__ void norms_kernel(const float* __restrict__ feat, float* __restrict__ norms,
                             int rows, int c) {
  int r = blockIdx.x * blockDim.x + threadIdx.x;
  if (r >= rows) return;
  float s = 0.f;
  for (int ch = 0; ch < c; ch++) { float v = feat[(size_t)r * c + ch]; s += v * v; }
  norms[r] = s;
}

__global__ void copy_chan_kernel(const float* __restrict__ src, float* __restrict__ dst,
                                 int rows, int cs, int cd, int off) {
  int t = blockIdx.x * blockDim.x + threadIdx.x;
  if (t >= rows * cs) return;
  int r = t / cs, ch = t % cs;
  dst[(size_t)r * cd + off + ch] = src[t];
}

__global__ void concat4_kernel(const float* __restrict__ x1, const float* __restrict__ x2,
                               const float* __restrict__ x3, const float* __restrict__ x4,
                               _Float16* __restrict__ out, int rows) {
  int t = blockIdx.x * blockDim.x + threadIdx.x;
  if (t >= rows * 512) return;
  int r = t / 512, ch = t % 512;
  float v;
  if (ch < 64)       v = x1[(size_t)r * 64 + ch];
  else if (ch < 128) v = x2[(size_t)r * 64 + (ch - 64)];
  else if (ch < 256) v = x3[(size_t)r * 128 + (ch - 128)];
  else               v = x4[(size_t)r * 256 + (ch - 256)];
  out[t] = (_Float16)v;
}

// ---------------------------------------------------------------------------
// kNN: gram matrix via WMMA  (neg_d = 2*x_i.x_j - ||x_i||^2 - ||x_j||^2)
// Each wave caches one A panel (16 rows) and sweeps 8 column tiles.
// grid = N/16 * 8 / 8 waves = 64 blocks of 256 threads (per batch).
// ---------------------------------------------------------------------------
template <int CP>
__global__ void gram_knn_kernel(const _Float16* __restrict__ xh, const float* __restrict__ norms,
                                float* __restrict__ negd, int N) {
  const int waveId = blockIdx.x * 8 + (threadIdx.x >> 5);
  const int rt = waveId >> 3, cg = waveId & 7;
  const int lane = threadIdx.x & 31;
  const int half = lane >> 4, l16 = lane & 15;
  const _Float16* arow = xh + (size_t)(rt * 16 + l16) * CP;
  v16h afrag[CP / 32];
#pragma unroll
  for (int kf = 0; kf < CP / 32; kf++) afrag[kf] = loadA_contig(arow, kf * 32, half);
  float normi[8];
#pragma unroll
  for (int r = 0; r < 8; r++) normi[r] = norms[rt * 16 + r + 8 * half];
  for (int cc = 0; cc < 8; cc++) {
    const int ct = cg * 8 + cc;
    const _Float16* brow = xh + (size_t)(ct * 16 + l16) * CP;
    v8f acc = {};
#pragma unroll
    for (int kf = 0; kf < CP / 32; kf++)
      acc = wmma16(afrag[kf], load16h(brow + kf * 32 + half * 16), acc);
    const int j = ct * 16 + l16;
    const float xxj = norms[j];
#pragma unroll
    for (int r = 0; r < 8; r++) {
      const int i = rt * 16 + r + 8 * half;
      negd[(size_t)i * N + j] = 2.f * acc[r] - normi[r] - xxj;
    }
  }
}

// top-K largest with lax.top_k tie semantics (lower index first).
__global__ void topk_kernel(const float* __restrict__ negd, int* __restrict__ idx,
                            int N, int K) {
  int i = blockIdx.x * blockDim.x + threadIdx.x;
  if (i >= N) return;
  const float* row = negd + (size_t)i * N;
  float pv = 3.4e38f; int pj = -1;
  for (int t = 0; t < K; t++) {
    float bv = -3.4e38f; int bj = N;
    for (int j = 0; j < N; j++) {
      float v = row[j];
      bool lt_prev = (v < pv) || (v == pv && j > pj);
      bool gt_best = (v > bv) || (v == bv && j < bj);
      if (lt_prev && gt_best) { bv = v; bj = j; }
    }
    idx[(size_t)i * K + t] = bj;
    pv = bv; pj = bj;
  }
}

// ---------------------------------------------------------------------------
// Fused edge-conv GEMM (ef = [nbr-ctr, ctr]) with two-pass batch-norm.
// mode 0: stats pass (per-workgroup sum/sumsq partials)
// mode 1: finalize (bn + leaky + max over k) -> xout (B,N,o)
// grid = B*N/16 workgroups of 256 threads (8 waves). Each wave builds the
// ef A-fragments for a row tile ONCE, then sweeps all output tiles.
// ---------------------------------------------------------------------------
#define ECP 16
#define ECK 20

template <int KP>
__global__ void edge_conv_kernel(const _Float16* __restrict__ xh, const int* __restrict__ nidx,
                                 const _Float16* __restrict__ wh,
                                 const float* __restrict__ gamma, const float* __restrict__ beta,
                                 const float* __restrict__ bnp,
                                 float* __restrict__ partial, float* __restrict__ xout,
                                 int N, int c, int cp, int o, int mode) {
  __shared__ float    s_stats[8 * 256 * 2];   // per-wave sum/sumsq, o <= 256
  __shared__ unsigned s_max[ECP * 256];       // sortable-uint running max
  const int wg = blockIdx.x;
  const int b  = wg / (N / ECP);
  const int pg = wg % (N / ECP);
  const int tid  = threadIdx.x;
  const int wave = tid >> 5, lane = tid & 31;
  const int half = lane >> 4, l16 = lane & 15;
  const bool vecOK = (c >= 8) && ((c & 7) == 0);

  if (mode == 0) { for (int t = tid; t < 8 * o * 2; t += blockDim.x) s_stats[t] = 0.f; }
  else           { for (int t = tid; t < ECP * o;  t += blockDim.x) s_max[t] = 0u; }
  __syncthreads();

  const int rts = (ECP * ECK) / 16;  // 20 row tiles
  const int ots = o / 16;

  for (int rt = wave; rt < rts; rt += 8) {
    const int row = rt * 16 + l16;                  // A-row of this lane
    const int pt = row / ECK, kk = row % ECK;
    const int n = pg * ECP + pt;
    const int nb = nidx[((size_t)b * N + n) * ECK + kk];
    const _Float16* ctrRow = xh + ((size_t)b * N + n)  * cp;
    const _Float16* nbrRow = xh + ((size_t)b * N + nb) * cp;

    // Build ef A-fragments once for this row tile.
    v16h afrag[KP / 32];
#pragma unroll
    for (int kf = 0; kf < KP / 32; kf++) {
      HalfPack u;
      u.f[0] = build_ef_run(nbrRow, ctrRow, kf * 32 + half * 16,     c, vecOK);
      u.f[1] = build_ef_run(nbrRow, ctrRow, kf * 32 + half * 16 + 8, c, vecOK);
      afrag[kf] = u.h;
    }

    for (int ct = 0; ct < ots; ct++) {
      const int och = ct * 16 + l16;
      const _Float16* wrow = wh + (size_t)och * KP;
      __builtin_prefetch(wrow, 0, 1);               // global_prefetch_b8
      v8f acc = {};
#pragma unroll
      for (int kf = 0; kf < KP / 32; kf++)
        acc = wmma16(afrag[kf], load16h(wrow + kf * 32 + half * 16), acc);

      if (mode == 0) {
        float s = 0.f, s2 = 0.f;
#pragma unroll
        for (int r = 0; r < 8; r++) { const float v = acc[r]; s += v; s2 += v * v; }
        const float so  = __shfl_down(s,  16);
        const float so2 = __shfl_down(s2, 16);
        if (half == 0) {
          const int si = (wave * o + och) * 2;
          s_stats[si]     += s  + so;
          s_stats[si + 1] += s2 + so2;
        }
      } else {
        const float mean = bnp[och * 2], inv = bnp[och * 2 + 1];
        const float g = gamma[och], bt = beta[och];
#pragma unroll
        for (int r = 0; r < 8; r++) {
          const int rrow = rt * 16 + r + 8 * half;
          const int ppt = rrow / ECK;
          float v = g * (acc[r] - mean) * inv + bt;
          v = (v >= 0.f) ? v : 0.2f * v;            // leaky
          atomicMax(&s_max[ppt * o + och], f2sort(v));
        }
      }
    }
  }
  __syncthreads();
  if (mode == 0) {
    for (int ch = tid; ch < o; ch += blockDim.x) {
      float s = 0.f, s2 = 0.f;
      for (int w = 0; w < 8; w++) {
        const int si = (w * o + ch) * 2;
        s += s_stats[si]; s2 += s_stats[si + 1];
      }
      partial[((size_t)wg * o + ch) * 2]     = s;
      partial[((size_t)wg * o + ch) * 2 + 1] = s2;
    }
  } else {
    for (int t = tid; t < ECP * o; t += blockDim.x) {
      const int pt = t / o, ch = t % o;
      const int n = pg * ECP + pt;
      xout[((size_t)b * N + n) * o + ch] = sort2f(s_max[t]);
    }
  }
}

// ---------------------------------------------------------------------------
// Generic fused linear layer: out = act(bn(fh @ wh^T + bias))
// A fragments are built once per workgroup (row tile fixed) and reused.
// mode 0: stats pass (if bn), mode 1: finalize. grid = rows/16 workgroups.
// act: 0 none, 1 relu, 2 leaky
// ---------------------------------------------------------------------------
template <int KP>
__global__ void lin_kernel(const _Float16* __restrict__ fh, const _Float16* __restrict__ wh,
                           const float* __restrict__ bias,
                           const float* __restrict__ gamma, const float* __restrict__ beta,
                           const float* __restrict__ bnp,
                           float* __restrict__ partial, float* __restrict__ out,
                           int o, int mode, int act, int hasBias, int hasBN) {
  __shared__ float s_stats[8 * 512 * 2];      // o <= 512
  const int wg = blockIdx.x;
  const int tid  = threadIdx.x;
  const int wave = tid >> 5, lane = tid & 31;
  const int half = lane >> 4, l16 = lane & 15;
  const int i0 = wg * 16;
  if (mode == 0) {
    for (int t = tid; t < 8 * o * 2; t += blockDim.x) s_stats[t] = 0.f;
    __syncthreads();
  }
  const int ots = o / 16;
  const _Float16* arow = fh + (size_t)(i0 + l16) * KP;
  v16h afrag[KP / 32];
#pragma unroll
  for (int kf = 0; kf < KP / 32; kf++) afrag[kf] = loadA_contig(arow, kf * 32, half);

  for (int job = wave; job < ots; job += 8) {
    const int och = job * 16 + l16;
    const _Float16* wrow = wh + (size_t)och * KP;
    __builtin_prefetch(wrow, 0, 1);
    v8f acc = {};
#pragma unroll
    for (int kf = 0; kf < KP / 32; kf++)
      acc = wmma16(afrag[kf], load16h(wrow + kf * 32 + half * 16), acc);
    const float bsv = hasBias ? bias[och] : 0.f;
    if (mode == 0) {
      float s = 0.f, s2 = 0.f;
#pragma unroll
      for (int r = 0; r < 8; r++) { const float v = acc[r] + bsv; s += v; s2 += v * v; }
      const float so  = __shfl_down(s,  16);
      const float so2 = __shfl_down(s2, 16);
      if (half == 0) {
        const int si = (wave * o + och) * 2;
        s_stats[si]     += s  + so;
        s_stats[si + 1] += s2 + so2;
      }
    } else {
      float mean = 0.f, inv = 1.f, g = 1.f, bt = 0.f;
      if (hasBN) { mean = bnp[och * 2]; inv = bnp[och * 2 + 1]; g = gamma[och]; bt = beta[och]; }
#pragma unroll
      for (int r = 0; r < 8; r++) {
        float v = acc[r] + bsv;
        if (hasBN) v = g * (v - mean) * inv + bt;
        if (act == 1)      v = fmaxf(v, 0.f);
        else if (act == 2) v = (v >= 0.f) ? v : 0.2f * v;
        const int rrow = i0 + r + 8 * half;
        out[(size_t)rrow * o + och] = v;
      }
    }
  }
  if (mode == 0) {
    __syncthreads();
    for (int ch = tid; ch < o; ch += blockDim.x) {
      float s = 0.f, s2 = 0.f;
      for (int w = 0; w < 8; w++) {
        const int si = (w * o + ch) * 2;
        s += s_stats[si]; s2 += s_stats[si + 1];
      }
      partial[((size_t)wg * o + ch) * 2]     = s;
      partial[((size_t)wg * o + ch) * 2 + 1] = s2;
    }
  }
}

__global__ void bn_reduce_kernel(const float* __restrict__ partial, int nwg, int o,
                                 float count, float* __restrict__ bnp) {
  int ch = blockIdx.x * blockDim.x + threadIdx.x;
  if (ch >= o) return;
  float s = 0.f, s2 = 0.f;
  for (int w = 0; w < nwg; w++) {
    s  += partial[((size_t)w * o + ch) * 2];
    s2 += partial[((size_t)w * o + ch) * 2 + 1];
  }
  float m = s / count;
  float var = s2 / count - m * m;
  bnp[ch * 2]     = m;
  bnp[ch * 2 + 1] = rsqrtf(var + 1e-5f);
}

// ---------------------------------------------------------------------------
// adapt_layer components (scalar; tiny FLOPs)
// ---------------------------------------------------------------------------
__global__ void fps_kernel(const float* __restrict__ pts, int* __restrict__ fidx,
                           int N, int S) {
  __shared__ float dist[1024];
  __shared__ float rv[256];
  __shared__ int   ri[256];
  __shared__ int   s_far;
  const int b = blockIdx.x, tid = threadIdx.x;
  const float* xyz = pts + (size_t)b * N * 3;
  for (int i = tid; i < N; i += 256) dist[i] = 1e10f;
  if (tid == 0) s_far = 0;
  __syncthreads();
  for (int t = 0; t < S; t++) {
    const int far = s_far;
    if (tid == 0) fidx[b * S + t] = far;
    const float cx = xyz[far * 3], cy = xyz[far * 3 + 1], cz = xyz[far * 3 + 2];
    float bv = -1.f; int bi = 0;
    for (int i = tid; i < N; i += 256) {
      float dx = xyz[i * 3] - cx, dy = xyz[i * 3 + 1] - cy, dz = xyz[i * 3 + 2] - cz;
      float nd = fminf(dist[i], dx * dx + dy * dy + dz * dz);
      dist[i] = nd;
      if (nd > bv || (nd == bv && i < bi)) { bv = nd; bi = i; }
    }
    rv[tid] = bv; ri[tid] = bi;
    __syncthreads();
    for (int s = 128; s > 0; s >>= 1) {
      if (tid < s) {
        if (rv[tid + s] > rv[tid] || (rv[tid + s] == rv[tid] && ri[tid + s] < ri[tid])) {
          rv[tid] = rv[tid + s]; ri[tid] = ri[tid + s];
        }
      }
      __syncthreads();
    }
    if (tid == 0) s_far = ri[0];
    __syncthreads();
  }
}

__global__ void gather_loc_kernel(const float* __restrict__ pts, const int* __restrict__ fidx,
                                  float* __restrict__ fploc, int B, int N, int S) {
  int t = blockIdx.x * blockDim.x + threadIdx.x;
  if (t >= B * S) return;
  int b = t / S;
  int fi = fidx[t];
  for (int d = 0; d < 3; d++) fploc[t * 3 + d] = pts[((size_t)b * N + fi) * 3 + d];
}

__global__ void ball_query_kernel(const float* __restrict__ pts, const float* __restrict__ centers,
                                  int* __restrict__ gidx, int B, int N, int S, int ns, float r2) {
  int t = blockIdx.x * blockDim.x + threadIdx.x;
  if (t >= B * S) return;
  int b = t / S;
  const float* xyz = pts + (size_t)b * N * 3;
  const float cx = centers[t * 3], cy = centers[t * 3 + 1], cz = centers[t * 3 + 2];
  int* out = gidx + (size_t)t * ns;
  int cnt = 0, first = -1;
  for (int j = 0; j < N && cnt < ns; j++) {
    float dx = xyz[j * 3] - cx, dy = xyz[j * 3 + 1] - cy, dz = xyz[j * 3 + 2] - cz;
    if (dx * dx + dy * dy + dz * dz <= r2) { if (first < 0) first = j; out[cnt++] = j; }
  }
  if (first < 0) first = N - 1;              // all out of range -> min(N, N-1)
  for (; cnt < ns; cnt++) out[cnt] = first;
}

__global__ void node_off_kernel(const float* __restrict__ fea, const float* __restrict__ pts,
                                const int* __restrict__ fidx, const int* __restrict__ gidx,
                                const float* __restrict__ wo,
                                const float* __restrict__ fploc, float* __restrict__ node_loc,
                                int B, int N, int S, int ns) {
  int t = blockIdx.x * blockDim.x + threadIdx.x;
  if (t >= B * S) return;
  int b = t / S;
  int fi = fidx[t];
  const float* ffea = fea + ((size_t)b * N + fi) * 64;
  const float* floc = fploc + (size_t)t * 3;
  float off0 = 0.f, off1 = 0.f, off2 = 0.f;
  for (int n = 0; n < ns; n++) {
    int j = gidx[(size_t)t * ns + n];
    const float* jf = fea + ((size_t)b * N + j) * 64;
    float d0 = 0.f, d1 = 0.f, d2 = 0.f;
    for (int ch = 0; ch < 64; ch++) {
      float g = jf[ch] - ffea[ch];
      d0 += g * wo[ch]; d1 += g * wo[64 + ch]; d2 += g * wo[128 + ch];
    }
    const float* jl = pts + ((size_t)b * N + j) * 3;
    off0 += tanhf(d0) * (jl[0] - floc[0]);
    off1 += tanhf(d1) * (jl[1] - floc[1]);
    off2 += tanhf(d2) * (jl[2] - floc[2]);
  }
  node_loc[t * 3]     = floc[0] + off0 / ns;
  node_loc[t * 3 + 1] = floc[1] + off1 / ns;
  node_loc[t * 3 + 2] = floc[2] + off2 / ns;
}

__global__ void node_fea_kernel(const float* __restrict__ res, const int* __restrict__ gidx2,
                                float* __restrict__ node_fea, int B, int N, int S, int ns, int C) {
  int t = blockIdx.x * blockDim.x + threadIdx.x;
  if (t >= B * S * C) return;
  int b = t / (S * C);
  int s = (t / C) % S;
  int ch = t % C;
  float m = -3.4e38f;
  for (int n = 0; n < ns; n++) {
    int j = gidx2[((size_t)(b * S + s)) * ns + n];
    m = fmaxf(m, res[((size_t)b * N + j) * C + ch]);
  }
  node_fea[t] = m;
}

__global__ void interp_kernel(const float* __restrict__ pts, const float* __restrict__ node_loc,
                              const float* __restrict__ node_fea, float* __restrict__ out,
                              int B, int N, int S, int C, int stride, int chOff) {
  int t = blockIdx.x * blockDim.x + threadIdx.x;
  if (t >= B * N) return;
  int b = t / N;
  const float px = pts[t * 3], py = pts[t * 3 + 1], pz = pts[t * 3 + 2];
  float w[3]; int id3[3];
  float pv = 3.4e38f; int pj = -1;
  for (int k = 0; k < 3; k++) {
    float bv = -3.4e38f; int bj = S;
    for (int j = 0; j < S; j++) {
      const float* nl = node_loc + ((size_t)(b * S + j)) * 3;
      float dx = px - nl[0], dy = py - nl[1], dz = pz - nl[2];
      float v = -(dx * dx + dy * dy + dz * dz);
      bool lt_prev = (v < pv) || (v == pv && j > pj);
      bool gt_best = (v > bv) || (v == bv && j < bj);
      if (lt_prev && gt_best) { bv = v; bj = j; }
    }
    id3[k] = bj;
    w[k] = 1.f / fmaxf(-bv, 1e-10f);
    pv = bv; pj = bj;
  }
  const float wsum = w[0] + w[1] + w[2];
  for (int ch = 0; ch < C; ch++) {
    float v = node_fea[((size_t)(b * S + id3[0])) * C + ch] * w[0] +
              node_fea[((size_t)(b * S + id3[1])) * C + ch] * w[1] +
              node_fea[((size_t)(b * S + id3[2])) * C + ch] * w[2];
    out[(size_t)t * stride + chOff + ch] = v / wsum;
  }
}

// ---------------------------------------------------------------------------
// Final reductions / outputs
// ---------------------------------------------------------------------------
__global__ void final_out_kernel(const float* __restrict__ x5, float* __restrict__ out,
                                 int B, int N, int C) {
  int t = blockIdx.x * blockDim.x + threadIdx.x;
  if (t >= B * C) return;
  int b = t / C, ch = t % C;
  float m = -3.4e38f, s = 0.f;
  for (int n = 0; n < N; n++) {
    float v = x5[((size_t)b * N + n) * C + ch];
    m = fmaxf(m, v); s += v;
  }
  out[(size_t)b * 2 * C + ch]     = m;
  out[(size_t)b * 2 * C + C + ch] = s / N;
}

__global__ void nodefea_out_kernel(const float* __restrict__ node_fea, float* __restrict__ out,
                                   int B, int S, int C) {
  int t = blockIdx.x * blockDim.x + threadIdx.x;
  if (t >= B * S * C) return;
  int b = t / (C * S);
  int c = (t / S) % C;
  int s = t % S;
  out[t] = node_fea[((size_t)(b * S + s)) * C + c];   // (B, C, S, 1)
}

// ---------------------------------------------------------------------------
// Host orchestration
// ---------------------------------------------------------------------------
extern "C" void kernel_launch(void* const* d_in, const int* in_sizes, int n_in,
                              void* d_out, int out_size, void* d_ws, size_t ws_size,
                              hipStream_t stream) {
  (void)in_sizes; (void)n_in; (void)out_size; (void)ws_size;
  const int B = 8, N = 1024, KNN = 20, S = 64, NS = 64;
  const int TPB = 256;

  const float* x   = (const float*)d_in[0];
  const float* w1  = (const float*)d_in[1];
  const float* g1  = (const float*)d_in[2];
  const float* be1 = (const float*)d_in[3];
  const float* w2  = (const float*)d_in[4];
  const float* g2  = (const float*)d_in[5];
  const float* be2 = (const float*)d_in[6];
  const float* w3  = (const float*)d_in[7];
  const float* g3  = (const float*)d_in[8];
  const float* be3 = (const float*)d_in[9];
  const float* w4  = (const float*)d_in[10];
  const float* g4  = (const float*)d_in[11];
  const float* be4 = (const float*)d_in[12];
  const float* w5  = (const float*)d_in[13];
  const float* g5  = (const float*)d_in[14];
  const float* be5 = (const float*)d_in[15];
  const float* wr  = (const float*)d_in[16];
  const float* br  = (const float*)d_in[17];
  const float* gr  = (const float*)d_in[18];
  const float* btr = (const float*)d_in[19];
  const float* wo  = (const float*)d_in[20];
  const float* wc  = (const float*)d_in[21];
  const float* bc  = (const float*)d_in[22];

  char* wsb = (char*)d_ws;
  size_t off = 0;
  auto alloc = [&](size_t bytes) -> void* {
    void* p = wsb + off;
    off += (bytes + 255) & ~(size_t)255;
    return p;
  };
  float*    pts     = (float*)alloc((size_t)B * N * 3 * 4);
  float*    negd    = (float*)alloc((size_t)N * N * 4);          // per-batch reuse
  int*      idx     = (int*)alloc((size_t)B * N * KNN * 4);
  float*    norms   = (float*)alloc((size_t)B * N * 4);
  _Float16* xh      = (_Float16*)alloc((size_t)B * N * 512 * 2); // f16 feature staging
  _Float16* wh      = (_Float16*)alloc((size_t)512 * 512 * 2);   // f16 weight staging
  float*    x1      = (float*)alloc((size_t)B * N * 64 * 4);
  float*    x2e     = (float*)alloc((size_t)B * N * 64 * 4);
  float*    xmid    = (float*)alloc((size_t)B * N * 128 * 4);    // x_ = [x2e, interp]
  float*    x2      = (float*)alloc((size_t)B * N * 64 * 4);
  float*    x3      = (float*)alloc((size_t)B * N * 128 * 4);
  float*    x4      = (float*)alloc((size_t)B * N * 256 * 4);
  float*    x5      = (float*)alloc((size_t)B * N * 512 * 4);
  float*    partial = (float*)alloc((size_t)512 * 512 * 2 * 4);
  float*    bnp     = (float*)alloc((size_t)512 * 2 * 4);
  int*      fidx    = (int*)alloc((size_t)B * S * 4);
  float*    fploc   = (float*)alloc((size_t)B * S * 3 * 4);
  float*    nloc    = (float*)alloc((size_t)B * S * 3 * 4);
  int*      gidx    = (int*)alloc((size_t)B * S * NS * 4);
  int*      gidx2   = (int*)alloc((size_t)B * S * NS * 4);
  float*    resb    = (float*)alloc((size_t)B * N * 64 * 4);
  float*    nfea    = (float*)alloc((size_t)B * S * 64 * 4);

  auto nb = [](long total, int bs) { return (unsigned)((total + bs - 1) / bs); };

  extract_pts_kernel<<<nb((long)B * N * 3, TPB), TPB, 0, stream>>>(x, pts, B, N);

  auto to16 = [&](const float* src, _Float16* dst, int rows, int c, int cp) {
    f32_to_f16_pad_kernel<<<nb((long)rows * cp, TPB), TPB, 0, stream>>>(src, dst, rows, c, cp);
  };

  auto run_knn = [&](const float* feat, int c, int cp) {
    to16(feat, xh, B * N, c, cp);
    norms_kernel<<<nb((long)B * N, TPB), TPB, 0, stream>>>(feat, norms, B * N, c);
    for (int b = 0; b < B; b++) {
      const _Float16* xb = xh + (size_t)b * N * cp;
      const float* nr = norms + (size_t)b * N;
      if (cp == 32)
        gram_knn_kernel<32><<<dim3(N / 16 * 8 / 8), dim3(TPB), 0, stream>>>(xb, nr, negd, N);
      else if (cp == 64)
        gram_knn_kernel<64><<<dim3(N / 16 * 8 / 8), dim3(TPB), 0, stream>>>(xb, nr, negd, N);
      else
        gram_knn_kernel<128><<<dim3(N / 16 * 8 / 8), dim3(TPB), 0, stream>>>(xb, nr, negd, N);
      topk_kernel<<<nb((long)N, TPB), TPB, 0, stream>>>(negd, idx + (size_t)b * N * KNN, N, KNN);
    }
  };

  auto run_edge = [&](const float* wsrc, const float* gamma, const float* beta,
                      int c, int cp, int kp, int o, float* xout) {
    to16(wsrc, wh, o, 2 * c, kp);
    auto launch = [&](int mode, const float* ga, const float* be, float* po, float* xo) {
      if (kp == 32)
        edge_conv_kernel<32><<<dim3(B * N / 16), dim3(TPB), 0, stream>>>(
            xh, idx, wh, ga, be, bnp, po, xo, N, c, cp, o, mode);
      else if (kp == 128)
        edge_conv_kernel<128><<<dim3(B * N / 16), dim3(TPB), 0, stream>>>(
            xh, idx, wh, ga, be, bnp, po, xo, N, c, cp, o, mode);
      else
        edge_conv_kernel<256><<<dim3(B * N / 16), dim3(TPB), 0, stream>>>(
            xh, idx, wh, ga, be, bnp, po, xo, N, c, cp, o, mode);
    };
    launch(0, nullptr, nullptr, partial, nullptr);
    bn_reduce_kernel<<<nb((long)o, 64), 64, 0, stream>>>(partial, B * N / 16, o,
                                                         (float)((long)B * N * KNN), bnp);
    launch(1, gamma, beta, nullptr, xout);
  };

  auto run_lin = [&](int kp, int o, const float* bias, int hasBias, int hasBN,
                     const float* gamma, const float* beta, int act, float* out) {
    auto launch = [&](int mode, float* po, float* xo) {
      if (kp == 64)
        lin_kernel<64><<<dim3(B * N / 16), dim3(TPB), 0, stream>>>(
            xh, wh, bias, gamma, beta, bnp, po, xo, o, mode, act, hasBias, hasBN);
      else if (kp == 128)
        lin_kernel<128><<<dim3(B * N / 16), dim3(TPB), 0, stream>>>(
            xh, wh, bias, gamma, beta, bnp, po, xo, o, mode, act, hasBias, hasBN);
      else
        lin_kernel<512><<<dim3(B * N / 16), dim3(TPB), 0, stream>>>(
            xh, wh, bias, gamma, beta, bnp, po, xo, o, mode, act, hasBias, hasBN);
    };
    if (hasBN) {
      launch(0, partial, nullptr);
      bn_reduce_kernel<<<nb((long)o, 64), 64, 0, stream>>>(partial, B * N / 16, o,
                                                           (float)(B * N), bnp);
    }
    launch(1, nullptr, out);
  };

  // ---- edge_conv 1 (pts, c=3) ----
  run_knn(pts, 3, 32);
  run_edge(w1, g1, be1, 3, 32, 32, 64, x1);

  // ---- edge_conv 2 (x1, c=64) ----
  run_knn(x1, 64, 64);
  run_edge(w2, g2, be2, 64, 64, 128, 64, x2e);

  // ---- adapt_layer ----
  fps_kernel<<<dim3(B), dim3(256), 0, stream>>>(pts, fidx, N, S);
  gather_loc_kernel<<<nb((long)B * S, TPB), TPB, 0, stream>>>(pts, fidx, fploc, B, N, S);
  ball_query_kernel<<<nb((long)B * S, TPB), TPB, 0, stream>>>(pts, fploc, gidx, B, N, S, NS, 0.09f);
  node_off_kernel<<<nb((long)B * S, TPB), TPB, 0, stream>>>(x2e, pts, fidx, gidx, wo, fploc, nloc,
                                                            B, N, S, NS);
  ball_query_kernel<<<nb((long)B * S, TPB), TPB, 0, stream>>>(pts, nloc, gidx2, B, N, S, NS, 0.09f);

  // res = relu(bn(x2e @ wr^T + br))
  to16(x2e, xh, B * N, 64, 64);
  to16(wr, wh, 64, 64, 64);
  run_lin(64, 64, br, 1, 1, gr, btr, 1, resb);
  node_fea_kernel<<<nb((long)B * S * 64, TPB), TPB, 0, stream>>>(resb, gidx2, nfea, B, N, S, NS, 64);
  copy_chan_kernel<<<nb((long)B * N * 64, TPB), TPB, 0, stream>>>(x2e, xmid, B * N, 64, 128, 0);
  interp_kernel<<<nb((long)B * N, TPB), TPB, 0, stream>>>(pts, nloc, nfea, xmid, B, N, S, 64, 128, 64);

  // x2 = xmid @ wc^T + bc
  to16(xmid, xh, B * N, 128, 128);
  to16(wc, wh, 64, 128, 128);
  run_lin(128, 64, bc, 1, 0, nullptr, nullptr, 0, x2);

  // ---- edge_conv 3 (x2, c=64) ----
  run_knn(x2, 64, 64);
  run_edge(w3, g3, be3, 64, 64, 128, 128, x3);

  // ---- edge_conv 4 (x3, c=128) ----
  run_knn(x3, 128, 128);
  run_edge(w4, g4, be4, 128, 128, 256, 256, x4);

  // ---- x5 = leaky(bn(xcat @ w5^T)) ----
  concat4_kernel<<<nb((long)B * N * 512, TPB), TPB, 0, stream>>>(x1, x2, x3, x4, xh, B * N);
  to16(w5, wh, 512, 512, 512);
  run_lin(512, 512, nullptr, 0, 1, g5, be5, 2, x5);

  // ---- outputs ----
  float* outf = (float*)d_out;
  final_out_kernel<<<nb((long)B * 512, TPB), TPB, 0, stream>>>(x5, outf, B, N, 512);
  nodefea_out_kernel<<<nb((long)B * S * 64, TPB), TPB, 0, stream>>>(nfea, outf + B * 1024, B, S, 64);
}
